// GraphAttentionNetworks_20023137534835
// MI455X (gfx1250) — matrix operand order repaired and checked by reading, not compiled, not run on hardware
//
#include <hip/hip_runtime.h>
#include <hip/hip_bf16.h>

typedef __attribute__((ext_vector_type(16))) _Float16 v16h;
typedef __attribute__((ext_vector_type(8)))  float    v8f;

#define LR_ALPHA 0.02f

__device__ __forceinline__ float lrelu(float x) { return x >= 0.f ? x : LR_ALPHA * x; }

// Monotonic float -> uint encoding so unsigned atomicMax == float max.
__device__ __forceinline__ unsigned encf(float f) {
    unsigned u = __float_as_uint(f);
    return (u & 0x80000000u) ? ~u : (u | 0x80000000u);
}
__device__ __forceinline__ float decf(unsigned e) {
    unsigned u = (e & 0x80000000u) ? (e & 0x7fffffffu) : ~e;
    return __uint_as_float(u);
}

// ---------------------------------------------------------------------------
// Workspace layout (bytes, 256-aligned). Total ~12.2 MB.
// ---------------------------------------------------------------------------
constexpr size_t OFF_X     = 0;                         // x       : 16384*128 f32 = 8 MB
constexpr size_t OFF_WHT   = 8u * 1024 * 1024;          // Wh^T f16: 8*128*2048   = 4 MB
constexpr size_t OFF_WG16  = OFF_WHT  + 4u*1024*1024;   // Wg f16  : 128*128      = 32 KB
constexpr size_t OFF_E1    = OFF_WG16 + 64u*1024;       // e1      : 8*2048 f32
constexpr size_t OFF_E2    = OFF_E1   + 64u*1024;       // e2      : 8*2048 f32
constexpr size_t OFF_E2MAX = OFF_E2   + 64u*1024;       // e2max   : 8 f32
constexpr size_t OFF_POOL  = OFF_E2MAX + 256;           // pooled  : 8*128 u32 (encoded)

// ---------------------------------------------------------------------------
// K1: x = lrelu(layernorm(h @ W1^T + b1))   [16384 rows, 128 feats, K=10]
// One wave (32 lanes) per row; 4 feats per lane; shuffle reductions for LN.
// ---------------------------------------------------------------------------
__global__ void k_fc1_ln(const float* __restrict__ h, const float* __restrict__ W1,
                         const float* __restrict__ b1, float* __restrict__ X)
{
    __shared__ float sW1[1280];
    __shared__ float sb1[128];
    int t = threadIdx.x;
    for (int i = t; i < 1280; i += 256) sW1[i] = W1[i];
    if (t < 128) sb1[t] = b1[t];
    __syncthreads();

    int wave = t >> 5, lane = t & 31;
    int row  = blockIdx.x * 8 + wave;            // 2048 blocks * 8 rows
    const float* hr = h + (size_t)row * 10;
    float hk[10];
    #pragma unroll
    for (int k = 0; k < 10; ++k) hk[k] = hr[k];

    int d0 = lane * 4;
    float xv[4];
    #pragma unroll
    for (int q = 0; q < 4; ++q) {
        int d = d0 + q;
        float acc = sb1[d];
        #pragma unroll
        for (int k = 0; k < 10; ++k) acc += hk[k] * sW1[d * 10 + k];
        xv[q] = acc;
    }
    float s = xv[0] + xv[1] + xv[2] + xv[3];
    #pragma unroll
    for (int off = 16; off > 0; off >>= 1) s += __shfl_xor(s, off, 32);
    float mean = s * (1.0f / 128.0f);
    float v = 0.f;
    #pragma unroll
    for (int q = 0; q < 4; ++q) { float dl = xv[q] - mean; v += dl * dl; }
    #pragma unroll
    for (int off = 16; off > 0; off >>= 1) v += __shfl_xor(v, off, 32);
    float rstd = rsqrtf(fmaxf(v * (1.0f / 128.0f), 0.f) + 1e-5f);

    float4 o;
    o.x = lrelu((xv[0] - mean) * rstd);
    o.y = lrelu((xv[1] - mean) * rstd);
    o.z = lrelu((xv[2] - mean) * rstd);
    o.w = lrelu((xv[3] - mean) * rstd);
    *(float4*)(X + (size_t)row * 128 + d0) = o;
}

// ---------------------------------------------------------------------------
// K2: Wg f32 -> f16
// ---------------------------------------------------------------------------
__global__ void k_cvt_wg(const float* __restrict__ Wg, _Float16* __restrict__ Wg16)
{
    int i = blockIdx.x * 256 + threadIdx.x;     // 64 blocks * 256 = 16384
    Wg16[i] = (_Float16)Wg[i];
}

// ---------------------------------------------------------------------------
// K3: Wh = x @ Wg^T + bg via v_wmma_f32_16x16x32_f16.
// Block = 256 threads = 8 waves; 16 rows x 128 feats per block.
// Full 16x128 A tile staged once (single barrier), then 4 back-to-back WMMAs.
// Emits Wh^T (f16, [b][feat][node]) + e1 = Wh@a1, e2 = Wh@a2.
// ---------------------------------------------------------------------------
__global__ void k_gemm_wh(const float* __restrict__ X, const _Float16* __restrict__ Wg16,
                          const float* __restrict__ bg, const float* __restrict__ a1,
                          const float* __restrict__ a2, _Float16* __restrict__ WhT,
                          float* __restrict__ e1, float* __restrict__ e2)
{
    __shared__ alignas(32) _Float16 sA[2048];   // 16x128 A tile, 4 ISA fragments
    __shared__ float sWh[2048];                 // 16x128 result tile (f32)

    int t = threadIdx.x;
    int rowbase = blockIdx.x * 16;              // 1024 blocks
    int b = rowbase >> 11;
    int nbase = rowbase & 2047;
    int lane = t & 31, wave = t >> 5;
    int hi = lane >> 4, nn = lane & 15;
    int d = wave * 16 + nn;                     // output feature this lane owns
    int i = t >> 4;                             // row this thread stages (A tile)
    int kl = (t & 15) * 2;                      // k_local pair
    int LN = ((kl >> 4) << 4) | i;              // A-frag lane for (i, K=kl)
    int vv = kl & 15;                           // A-frag element index

    #pragma unroll
    for (int ic = 0; ic < 4; ++ic) {
        float2 xv = *(const float2*)(X + (size_t)(rowbase + i) * 128 + ic * 32 + kl);
        sA[ic * 512 + LN * 16 + vv]     = (_Float16)xv.x;
        sA[ic * 512 + LN * 16 + vv + 1] = (_Float16)xv.y;
    }
    __syncthreads();

    v8f c = {0.f, 0.f, 0.f, 0.f, 0.f, 0.f, 0.f, 0.f};
    #pragma unroll
    for (int ic = 0; ic < 4; ++ic) {
        v16h af = *(const v16h*)(sA + ic * 512 + lane * 16);
        v16h bf = *(const v16h*)(Wg16 + d * 128 + ic * 32 + hi * 16);
        c = __builtin_amdgcn_wmma_f32_16x16x32_f16(false, af, false, bf,
                                                   (short)0, c, false, false);
    }

    float bgv = bg[d];
    #pragma unroll
    for (int r = 0; r < 8; ++r) sWh[(r + hi * 8) * 128 + d] = c[r] + bgv;
    __syncthreads();

    if (t < 16) {
        float s1 = 0.f, s2 = 0.f;
        for (int dd = 0; dd < 128; ++dd) {
            float w = sWh[t * 128 + dd];
            s1 += w * a1[dd];
            s2 += w * a2[dd];
        }
        e1[b * 2048 + nbase + t] = s1;
        e2[b * 2048 + nbase + t] = s2;
    }

    // Wh^T f16: thread t packs 8 consecutive nodes for feature (t&127).
    int dd = t & 127, half = t >> 7;
    union { _Float16 hh[8]; uint4 u; } pk;
    #pragma unroll
    for (int q = 0; q < 8; ++q) pk.hh[q] = (_Float16)sWh[(half * 8 + q) * 128 + dd];
    *(uint4*)(WhT + (size_t)(b * 128 + dd) * 2048 + nbase + half * 8) = pk.u;
}

// ---------------------------------------------------------------------------
// K4: per-batch max(e2) (lrelu-monotonicity gives the softmax upper bound
// m_i = lrelu(e1_i + e2max) without touching adj), and init pooled maxes.
// ---------------------------------------------------------------------------
__global__ void k_rowstats(const float* __restrict__ e2, float* __restrict__ e2max,
                           unsigned* __restrict__ pool)
{
    __shared__ float sred[256];
    int b = blockIdx.x, t = threadIdx.x;
    float m = -3.0e38f;
    #pragma unroll
    for (int q = 0; q < 8; ++q) m = fmaxf(m, e2[b * 2048 + t * 8 + q]);
    sred[t] = m; __syncthreads();
    for (int s = 128; s > 0; s >>= 1) {
        if (t < s) sred[t] = fmaxf(sred[t], sred[t + s]);
        __syncthreads();
    }
    if (t == 0) e2max[b] = sred[0];
    if (t < 128) pool[b * 128 + t] = 0u;       // enc of any real value beats 0
}

// ---------------------------------------------------------------------------
// K5: masked softmax + P@Wh (WMMA) + layernorm + lrelu + max-pool.
// One pass over adj (134 MB, the roofline bound). Block = 16 query rows,
// 8 waves x 16 feats. 64 j per iteration, double-buffered P tile ->
// single barrier per iteration; 2 WMMAs per barrier.
// ---------------------------------------------------------------------------
__global__ void k_attn(const int* __restrict__ adj, const float* __restrict__ e1g,
                       const float* __restrict__ e2g, const float* __restrict__ e2max,
                       const _Float16* __restrict__ WhT, unsigned* __restrict__ pool)
{
    __shared__ alignas(32) _Float16 sP[2048];   // 2 x (16x64) P tiles, A-frag layout
    __shared__ float sBuf[2048];                // e2 row during loop; hp tile after
    __shared__ float sE1[16], sM[16];
    __shared__ float sred[256];
    __shared__ float sinvl[16], smu[16], srstd[16];

    int t = threadIdx.x;
    int rowbase = blockIdx.x * 16;              // 1024 blocks
    int b = rowbase >> 11;
    int ibase = rowbase & 2047;

    // Stage e2 row (8 KB) and per-row softmax bounds.
    {
        const float* e2row = e2g + b * 2048;
        #pragma unroll
        for (int q = 0; q < 2; ++q)
            *(float4*)(sBuf + t * 4 + q * 1024) = *(const float4*)(e2row + t * 4 + q * 1024);
    }
    if (t < 16) {
        float e1v = e1g[b * 2048 + ibase + t];
        sE1[t] = e1v;
        sM[t]  = lrelu(e1v + e2max[b]);         // >= all unmasked e in the row
    }
    __syncthreads();

    int lane = t & 31, wave = t >> 5;
    int hi = lane >> 4, nn = lane & 15;
    int d = wave * 16 + nn;
    int i = t >> 4;
    int j0 = (t & 15) * 4;                      // 4 consecutive j per thread
    int K  = j0 & 31;
    int LN = ((K >> 4) << 4) | i;
    int vv = K & 15;
    int chunk = (j0 >> 5) * 512;                // which 16x32 sub-tile

    float e1i = sE1[i], mi = sM[i];
    const int*      adjRow = adj + (size_t)(b * 2048 + ibase + i) * 2048;
    const _Float16* bcol   = WhT + (size_t)(b * 128 + d) * 2048 + hi * 16;

    v8f c = {0.f, 0.f, 0.f, 0.f, 0.f, 0.f, 0.f, 0.f};
    float lsum = 0.f;

    for (int jc = 0; jc < 2048; jc += 64) {
        _Float16* sPb = sP + ((jc >> 6) & 1) * 1024;   // double buffer

        __builtin_prefetch(adjRow + jc + 128 + j0, 0, 0);  // global_prefetch_b8

        int4   av = *(const int4*)(adjRow + jc + j0);
        float4 ev = *(const float4*)(sBuf + jc + j0);
        float p0 = av.x ? __expf(lrelu(e1i + ev.x) - mi) : 0.f;
        float p1 = av.y ? __expf(lrelu(e1i + ev.y) - mi) : 0.f;
        float p2 = av.z ? __expf(lrelu(e1i + ev.z) - mi) : 0.f;
        float p3 = av.w ? __expf(lrelu(e1i + ev.w) - mi) : 0.f;
        lsum += (p0 + p1) + (p2 + p3);

        union { _Float16 hh[4]; uint2 u; } pk;
        pk.hh[0] = (_Float16)p0; pk.hh[1] = (_Float16)p1;
        pk.hh[2] = (_Float16)p2; pk.hh[3] = (_Float16)p3;
        *(uint2*)(sPb + chunk + LN * 16 + vv) = pk.u;

        __syncthreads();   // waves wait dscnt before signaling -> WAR safe

        v16h a0 = *(const v16h*)(sPb + lane * 16);
        v16h a1f = *(const v16h*)(sPb + 512 + lane * 16);
        v16h b0 = *(const v16h*)(bcol + jc);
        v16h b1f = *(const v16h*)(bcol + jc + 32);
        c = __builtin_amdgcn_wmma_f32_16x16x32_f16(false, a0, false, b0,
                                                   (short)0, c, false, false);
        c = __builtin_amdgcn_wmma_f32_16x16x32_f16(false, a1f, false, b1f,
                                                   (short)0, c, false, false);
    }

    // l_i reduction (16 threads per row are contiguous: t = i*16 + q).
    sred[t] = lsum; __syncthreads();
    if (t < 16) {
        float l = 0.f;
        #pragma unroll
        for (int q = 0; q < 16; ++q) l += sred[t * 16 + q];
        sinvl[t] = (l > 0.f) ? 1.0f / l : 0.f;
    }
    __syncthreads();

    // Normalize accumulators and stage hp tile (sBuf reused: e2 no longer read).
    #pragma unroll
    for (int r = 0; r < 8; ++r)
        sBuf[(r + hi * 8) * 128 + d] = c[r] * sinvl[r + hi * 8];
    __syncthreads();

    if (t < 16) {
        float s = 0.f;
        for (int dd = 0; dd < 128; ++dd) s += sBuf[t * 128 + dd];
        float mean = s * (1.0f / 128.0f);
        float v = 0.f;
        for (int dd = 0; dd < 128; ++dd) { float dl = sBuf[t * 128 + dd] - mean; v += dl * dl; }
        smu[t]   = mean;
        srstd[t] = rsqrtf(fmaxf(v * (1.0f / 128.0f), 0.f) + 1e-5f);
    }
    __syncthreads();

    if (t < 128) {
        float mx = -3.0e38f;
        #pragma unroll
        for (int q = 0; q < 16; ++q) {
            float val = (sBuf[q * 128 + t] - smu[q]) * srstd[q];
            mx = fmaxf(mx, lrelu(val));
        }
        atomicMax(pool + b * 128 + t, encf(mx));
    }
}

// ---------------------------------------------------------------------------
// K6: out = log_softmax(pooled @ W2^T + b2)   -> 8x2 floats
// ---------------------------------------------------------------------------
__global__ void k_head(const unsigned* __restrict__ pool, const float* __restrict__ W2,
                       const float* __restrict__ b2, float* __restrict__ out)
{
    int t = threadIdx.x;
    if (t < 8) {
        float z0 = b2[0], z1 = b2[1];
        for (int dd = 0; dd < 128; ++dd) {
            float p = decf(pool[t * 128 + dd]);
            z0 += p * W2[dd];
            z1 += p * W2[128 + dd];
        }
        float m = fmaxf(z0, z1);
        float lse = m + logf(__expf(z0 - m) + __expf(z1 - m));
        out[t * 2 + 0] = z0 - lse;
        out[t * 2 + 1] = z1 - lse;
    }
}

// ---------------------------------------------------------------------------
extern "C" void kernel_launch(void* const* d_in, const int* in_sizes, int n_in,
                              void* d_out, int out_size, void* d_ws, size_t ws_size,
                              hipStream_t stream)
{
    const float* h   = (const float*)d_in[0];
    const int*   adj = (const int*)  d_in[1];
    const float* W1  = (const float*)d_in[2];
    const float* b1  = (const float*)d_in[3];
    const float* Wg  = (const float*)d_in[4];
    const float* bg  = (const float*)d_in[5];
    const float* a1  = (const float*)d_in[6];
    const float* a2  = (const float*)d_in[7];
    const float* W2  = (const float*)d_in[8];
    const float* b2  = (const float*)d_in[9];
    float* out = (float*)d_out;

    char* ws = (char*)d_ws;
    float*     X     = (float*)    (ws + OFF_X);
    _Float16*  WhT   = (_Float16*) (ws + OFF_WHT);
    _Float16*  Wg16  = (_Float16*) (ws + OFF_WG16);
    float*     e1    = (float*)    (ws + OFF_E1);
    float*     e2    = (float*)    (ws + OFF_E2);
    float*     e2m   = (float*)    (ws + OFF_E2MAX);
    unsigned*  pool  = (unsigned*) (ws + OFF_POOL);

    k_fc1_ln  <<<2048, 256, 0, stream>>>(h, W1, b1, X);
    k_cvt_wg  <<<  64, 256, 0, stream>>>(Wg, Wg16);
    k_gemm_wh <<<1024, 256, 0, stream>>>(X, Wg16, bg, a1, a2, WhT, e1, e2);
    k_rowstats<<<   8, 256, 0, stream>>>(e2, e2m, pool);
    k_attn    <<<1024, 256, 0, stream>>>(adj, e1, e2, e2m, WhT, pool);
    k_head    <<<   1,  64, 0, stream>>>(pool, W2, b2, out);
}
